// PaTHAttentionAdapter_85237920957017
// MI455X (gfx1250) — compile-verified, matrix operand order checked by reading, not run on hardware
//
#include <hip/hip_runtime.h>
#include <math.h>

#define T_DIM 2048
#define D_DIM 1024
#define H_DIM 8
#define HDIM  128
#define BM 128
#define BN 128

typedef __attribute__((ext_vector_type(16))) _Float16 v16h;
typedef __attribute__((ext_vector_type(8)))  float    v8f;

enum { EPI_STORE=0, EPI_TRIL=1, EPI_TRILS=2, EPI_TRILS_CSCALE=3, EPI_NEG=4, EPI_SUB=5, EPI_RSCALE=6 };

// ---------------------------------------------------------------------------
// Generic WMMA GEMM: C = epilogue(A[M,K] x B). B is [K,N] row-major (BT=0,
// staged through LDS for coalescing) or [N,K] row-major, i.e. implicit
// transpose (BT=1, contiguous per-lane K runs straight from global/L2).
// f32 operands in memory, converted to f16 in registers, f32 accumulation via
// v_wmma_f32_16x16x32_f16. blockDim=256 (8 wave32s), block tile 128x128,
// wave tile 32x64 (2x4 WMMA tiles -> 8 WMMA per 6 fragment loads).
// TRIKLO/TRIKHI clamp the K loop for triangular operands.
// zs*: per-blockIdx.z offsets for the batched doubling GEMMs.
// ---------------------------------------------------------------------------
template<int EPI, bool BT, bool TRIKLO, bool TRIKHI>
__global__ __launch_bounds__(256) void gemm_wmma(
    const float* __restrict__ A, int lda,
    const float* __restrict__ B, int ldb,
    float* __restrict__ C, int ldc,
    int M, int N, int K,
    long long zsA, long long zsB, long long zsC,
    const float* __restrict__ sc, int scs)
{
  A += (long long)blockIdx.z * zsA;
  B += (long long)blockIdx.z * zsB;
  C += (long long)blockIdx.z * zsC;
  const int tid  = threadIdx.x;
  const int lane = tid & 31;
  const int wv   = tid >> 5;
  const int wm   = wv & 3;       // 4 waves along M
  const int wn   = wv >> 2;      // 2 waves along N
  const int m0   = blockIdx.y * BM;
  const int n0   = blockIdx.x * BN;
  const int wm0  = m0 + wm * 32;
  const int wn0  = n0 + wn * 64;
  const int hi   = lane >> 4;
  const int lo   = lane & 15;

  int kmin = 0, kmax = K;
  if (TRIKLO) { if (n0 > kmin) kmin = n0; }            // multiples of 128
  if (TRIKHI) { int v = m0 + BM; if (v < kmax) kmax = v; }

  v8f acc[2][4];
  #pragma unroll
  for (int i = 0; i < 2; i++)
    #pragma unroll
    for (int j = 0; j < 4; j++)
      #pragma unroll
      for (int e = 0; e < 8; e++) acc[i][j][e] = 0.0f;

  int rA[2], cB[4];
  #pragma unroll
  for (int i = 0; i < 2; i++) { int r = wm0 + i*16 + lo; rA[i] = (r < M) ? r : (M - 1); }
  #pragma unroll
  for (int j = 0; j < 4; j++) { int c = wn0 + j*16 + lo; cB[j] = (c < N) ? c : (N - 1); }

  for (int k0 = kmin; k0 < kmax; k0 += 32) {
    // prefetch next K tile (global_prefetch_b8)
    if (k0 + 32 < kmax) {
      __builtin_prefetch((const void*)(A + (long long)rA[0] * lda + (k0 + 32 + hi * 16)), 0, 1);
      if (BT)
        __builtin_prefetch((const void*)(B + (long long)cB[0] * ldb + (k0 + 32 + hi * 16)), 0, 1);
      else
        __builtin_prefetch((const void*)(B + (long long)(k0 + 32 + (tid >> 3)) * ldb + n0 + ((tid & 7) << 4)), 0, 1);
    }

    v16h af[2], bf[4];
    #pragma unroll
    for (int i = 0; i < 2; i++) {
      const float* p = A + (long long)rA[i] * lda + (k0 + hi * 16);
      #pragma unroll
      for (int e = 0; e < 16; e++) af[i][e] = (_Float16)p[e];
    }

    if constexpr (BT) {
      #pragma unroll
      for (int j = 0; j < 4; j++) {
        const float* p = B + (long long)cB[j] * ldb + (k0 + hi * 16);
        #pragma unroll
        for (int e = 0; e < 16; e++) bf[j][e] = (_Float16)p[e];
      }
    } else {
      // stage 32xBN B tile through LDS: coalesced global reads, padded rows
      // so the K-strided fragment reads are bank-conflict-free ds_load_b32.
      __shared__ float bs[32][BN + 1];
      __syncthreads();
      #pragma unroll
      for (int it = 0; it < (32 * BN) / 256; it++) {
        int idx = tid + it * 256;
        int kk = idx >> 7, nn = idx & (BN - 1);
        int gn = n0 + nn; if (gn >= N) gn = N - 1;
        bs[kk][nn] = B[(long long)(k0 + kk) * ldb + gn];
      }
      __syncthreads();
      #pragma unroll
      for (int j = 0; j < 4; j++) {
        int cl = cB[j] - n0;                  // local column (clamped already)
        #pragma unroll
        for (int e = 0; e < 16; e++) bf[j][e] = (_Float16)bs[hi * 16 + e][cl];
      }
    }

    #pragma unroll
    for (int i = 0; i < 2; i++)
      #pragma unroll
      for (int j = 0; j < 4; j++)
        acc[i][j] = __builtin_amdgcn_wmma_f32_16x16x32_f16(
            false, af[i], false, bf[j], (short)0, acc[i][j], false, false);
  }

  #pragma unroll
  for (int i = 0; i < 2; i++)
    #pragma unroll
    for (int r = 0; r < 8; r++) {
      int gr = wm0 + i * 16 + hi * 8 + r;
      if (gr < M) {
        float* crow = C + (long long)gr * ldc;
        #pragma unroll
        for (int j = 0; j < 4; j++) {
          int gc = wn0 + j * 16 + lo;
          if (gc < N) {
            float v = acc[i][j][r];
            if (EPI == EPI_TRIL)         v = (gc <= gr) ? v : 0.0f;
            if (EPI == EPI_TRILS)        v = (gc <  gr) ? v : 0.0f;
            if (EPI == EPI_TRILS_CSCALE) v = (gc <  gr) ? v * sc[(long long)gc * scs] : 0.0f;
            if (EPI == EPI_RSCALE)       v = v * sc[(long long)gr * scs];
            if (EPI == EPI_NEG)          v = -v;
            if (EPI == EPI_SUB) crow[gc] -= v; else crow[gc] = v;
          }
        }
      }
    }
}

// ---------------------------------------------------------------------------
// 1024x1024 transpose (weights -> W^T so big GEMMs get contiguous K loads)
// ---------------------------------------------------------------------------
__global__ void transpose_k(const float* __restrict__ in, float* __restrict__ out, int n)
{
  __shared__ float t[16][17];
  int bx = blockIdx.x * 16, by = blockIdx.y * 16;
  t[threadIdx.y][threadIdx.x] = in[(long long)(by + threadIdx.y) * n + bx + threadIdx.x];
  __syncthreads();
  out[(long long)(bx + threadIdx.y) * n + by + threadIdx.x] = t[threadIdx.x][threadIdx.y];
}

// ---------------------------------------------------------------------------
// beta = 2*sigmoid(x@Wbeta), g = logsigmoid(x@Wg)   (N=8, done in VALU)
// ---------------------------------------------------------------------------
__global__ __launch_bounds__(256) void betag_k(
    const float* __restrict__ x, const float* __restrict__ Wb,
    const float* __restrict__ Wg, float* __restrict__ beta, float* __restrict__ g)
{
  int t = blockIdx.x, tid = threadIdx.x;
  float ab[H_DIM], ag[H_DIM];
  #pragma unroll
  for (int h = 0; h < H_DIM; h++) { ab[h] = 0.f; ag[h] = 0.f; }
  for (int d = tid; d < D_DIM; d += 256) {
    float xv = x[(long long)t * D_DIM + d];
    #pragma unroll
    for (int h = 0; h < H_DIM; h++) {
      ab[h] += xv * Wb[d * H_DIM + h];
      ag[h] += xv * Wg[d * H_DIM + h];
    }
  }
  __shared__ float red[256];
  for (int h = 0; h < H_DIM; h++) {
    red[tid] = ab[h]; __syncthreads();
    for (int o = 128; o > 0; o >>= 1) { if (tid < o) red[tid] += red[tid + o]; __syncthreads(); }
    if (tid == 0) { float s = red[0]; beta[t * H_DIM + h] = 2.0f / (1.0f + expf(-s)); }
    __syncthreads();
    red[tid] = ag[h]; __syncthreads();
    for (int o = 128; o > 0; o >>= 1) { if (tid < o) red[tid] += red[tid + o]; __syncthreads(); }
    if (tid == 0) { float s = red[0]; g[t * H_DIM + h] = fminf(s, 0.0f) - log1pf(expf(-fabsf(s))); }
    __syncthreads();
  }
}

__global__ void cumsum_k(const float* __restrict__ g, float* __restrict__ G)
{
  int h = threadIdx.x;
  if (h >= H_DIM) return;
  float run = 0.f;
  for (int t = 0; t < T_DIM; t++) { run += g[t * H_DIM + h]; G[t * H_DIM + h] = run; }
}

// ---------------------------------------------------------------------------
// depthwise causal conv(K=4) + silu + per-head L2 normalize -> w
// ---------------------------------------------------------------------------
__global__ __launch_bounds__(128) void conv_silu_norm_k(
    const float* __restrict__ wpre, const float* __restrict__ cw, float* __restrict__ w)
{
  int t = blockIdx.x, h = blockIdx.y, d = threadIdx.x;
  int c = h * HDIM + d;
  float s = 0.f;
  #pragma unroll
  for (int i = 0; i < 4; i++) {
    int tt = t - 3 + i;
    if (tt >= 0) s += wpre[(long long)tt * D_DIM + c] * cw[c * 4 + i];
  }
  float val = s / (1.0f + expf(-s));   // silu
  __shared__ float red[128];
  red[d] = val * val; __syncthreads();
  for (int o = 64; o > 0; o >>= 1) { if (d < o) red[d] += red[d + o]; __syncthreads(); }
  w[(long long)t * D_DIM + c] = val * rsqrtf(red[0] + 1e-6f);
}

// per-head RMS norm (in place), weight length 128
__global__ __launch_bounds__(128) void rms_k(float* __restrict__ x, const float* __restrict__ wgt)
{
  int t = blockIdx.x, h = blockIdx.y, d = threadIdx.x;
  long long idx = (long long)t * D_DIM + h * HDIM + d;
  float v = x[idx];
  __shared__ float red[128];
  red[d] = v * v; __syncthreads();
  for (int o = 64; o > 0; o >>= 1) { if (d < o) red[d] += red[d + o]; __syncthreads(); }
  x[idx] = v * rsqrtf(red[0] * (1.0f / HDIM) + 1e-6f) * wgt[d];
}

// ---------------------------------------------------------------------------
// Invert each 64x64 unit-lower diagonal block of M (strict-lower stored),
// writing the full 64x64 inverse tile (zeros above diag) into Minv.
// ---------------------------------------------------------------------------
__global__ __launch_bounds__(64) void diaginv_k(
    const float* __restrict__ Mm, float* __restrict__ Minv, int T)
{
  int b0 = blockIdx.x * 64;
  int j = threadIdx.x;
  __shared__ float inv[64][65];
  for (int s = 0; s < 64; s++) inv[s][j] = (s == j) ? 1.0f : 0.0f;
  __syncthreads();
  for (int t = 1; t < 64; t++) {
    float a = 0.f;
    for (int s = 0; s < t; s++)
      a += Mm[(long long)(b0 + t) * T + b0 + s] * inv[s][j];
    inv[t][j] -= a;
    __syncthreads();
  }
  for (int s = 0; s < 64; s++)
    Minv[(long long)(b0 + s) * T + b0 + j] = inv[s][j];
}

// ---------------------------------------------------------------------------
// Row softmax over A with FoX gating; writes P in place (zeros for j > t).
// ---------------------------------------------------------------------------
__global__ __launch_bounds__(256) void softmax_k(
    float* __restrict__ Am, const float* __restrict__ G, int h, float scale)
{
  int t = blockIdx.x, tid = threadIdx.x;
  float Gt = G[t * H_DIM + h];
  __shared__ float red[256];
  float m = -3.0e38f;
  for (int j = tid; j <= t; j += 256) {
    float l = Am[(long long)t * T_DIM + j] * scale + Gt - G[j * H_DIM + h];
    m = fmaxf(m, l);
  }
  red[tid] = m; __syncthreads();
  for (int o = 128; o > 0; o >>= 1) { if (tid < o) red[tid] = fmaxf(red[tid], red[tid + o]); __syncthreads(); }
  m = red[0]; __syncthreads();
  float s = 0.f;
  for (int j = tid; j <= t; j += 256) {
    float l = Am[(long long)t * T_DIM + j] * scale + Gt - G[j * H_DIM + h];
    float p = expf(l - m);
    Am[(long long)t * T_DIM + j] = p;
    s += p;
  }
  red[tid] = s; __syncthreads();
  for (int o = 128; o > 0; o >>= 1) { if (tid < o) red[tid] += red[tid + o]; __syncthreads(); }
  float invs = 1.0f / red[0];
  for (int j = tid; j <= t; j += 256) Am[(long long)t * T_DIM + j] *= invs;
  for (int j = tid; j < T_DIM; j += 256) if (j > t) Am[(long long)t * T_DIM + j] = 0.0f;
}

// ---------------------------------------------------------------------------
extern "C" void kernel_launch(void* const* d_in, const int* in_sizes, int n_in,
                              void* d_out, int out_size, void* d_ws, size_t ws_size,
                              hipStream_t stream)
{
  (void)in_sizes; (void)n_in; (void)out_size; (void)ws_size;
  const int T = T_DIM, D = D_DIM, H = H_DIM;

  const float* x     = (const float*)d_in[0];
  const float* Wq    = (const float*)d_in[1];
  const float* Wk    = (const float*)d_in[2];
  const float* Wv    = (const float*)d_in[3];
  const float* Ww    = (const float*)d_in[4];
  const float* Wbeta = (const float*)d_in[5];
  const float* Wg    = (const float*)d_in[6];
  const float* Wo    = (const float*)d_in[7];
  const float* convw = (const float*)d_in[8];
  const float* qn_w  = (const float*)d_in[9];
  const float* kn_w  = (const float*)d_in[10];
  float* out = (float*)d_out;

  float* ws = (float*)d_ws;
  size_t off = 0;
  auto alloc = [&](size_t n) { float* p = ws + off; off += n; return p; };
  const size_t DD = (size_t)D * D, TD = (size_t)T * D, TT = (size_t)T * T;
  float* WqT = alloc(DD); float* WkT = alloc(DD); float* WvT = alloc(DD);
  float* WwT = alloc(DD); float* WoT = alloc(DD);
  float* q = alloc(TD); float* k = alloc(TD); float* v = alloc(TD);
  float* wpre = alloc(TD); float* w = alloc(TD); float* o = alloc(TD);
  float* beta = alloc((size_t)T * H); float* g = alloc((size_t)T * H); float* G = alloc((size_t)T * H);
  float* Mb   = alloc(TT);            // M strict-lower (later reused for tril(QW))
  float* Minv = alloc(TT);            // (I + L)^-1
  float* R    = alloc(TT);            // rhs, later scores A / probs P
  float* btil = alloc(TT);
  float* tmp  = alloc((size_t)(T / 2) * T);   // doubling scratch, ld = T

  dim3 tb(16, 16), tg(D / 16, D / 16);
  transpose_k<<<tg, tb, 0, stream>>>(Wq, WqT, D);
  transpose_k<<<tg, tb, 0, stream>>>(Wk, WkT, D);
  transpose_k<<<tg, tb, 0, stream>>>(Wv, WvT, D);
  transpose_k<<<tg, tb, 0, stream>>>(Ww, WwT, D);
  transpose_k<<<tg, tb, 0, stream>>>(Wo, WoT, D);

  // projections: [T,D] @ [D,D]
  dim3 gproj(D / BN, T / BM, 1);
  gemm_wmma<EPI_STORE, true, false, false><<<gproj, 256, 0, stream>>>(x, D, WqT, D, q,    D, T, D, D, 0, 0, 0, nullptr, 0);
  gemm_wmma<EPI_STORE, true, false, false><<<gproj, 256, 0, stream>>>(x, D, WkT, D, k,    D, T, D, D, 0, 0, 0, nullptr, 0);
  gemm_wmma<EPI_STORE, true, false, false><<<gproj, 256, 0, stream>>>(x, D, WvT, D, v,    D, T, D, D, 0, 0, 0, nullptr, 0);
  gemm_wmma<EPI_STORE, true, false, false><<<gproj, 256, 0, stream>>>(x, D, WwT, D, wpre, D, T, D, D, 0, 0, 0, nullptr, 0);

  betag_k<<<dim3(T), 256, 0, stream>>>(x, Wbeta, Wg, beta, g);
  cumsum_k<<<dim3(1), H, 0, stream>>>(g, G);
  conv_silu_norm_k<<<dim3(T, H), 128, 0, stream>>>(wpre, convw, w);
  rms_k<<<dim3(T, H), 128, 0, stream>>>(q, qn_w);
  rms_k<<<dim3(T, H), 128, 0, stream>>>(k, kn_w);

  const float scale = 1.0f / sqrtf((float)HDIM);
  dim3 gTT(T / BN, T / BM, 1);

  for (int h = 0; h < H; h++) {
    const float* qh = q + h * HDIM;
    const float* kh = k + h * HDIM;
    const float* vh = v + h * HDIM;
    const float* wh = w + h * HDIM;

    hipMemsetAsync(Minv, 0, TT * sizeof(float), stream);

    // Mstrict[t,s] = (w_t . w_s) * beta_s for s<t   ;  R[t,j] = w_t . k_j (strict lower)
    gemm_wmma<EPI_TRILS_CSCALE, true, false, false><<<gTT, 256, 0, stream>>>(wh, D, wh, D, Mb, T, T, T, HDIM, 0, 0, 0, beta + h, H);
    gemm_wmma<EPI_TRILS,        true, false, false><<<gTT, 256, 0, stream>>>(wh, D, kh, D, R,  T, T, T, HDIM, 0, 0, 0, nullptr, 0);

    // invert I + L by recursive doubling
    diaginv_k<<<dim3(T / 64), 64, 0, stream>>>(Mb, Minv, T);
    for (int c = 64; c <= T / 2; c *= 2) {
      int P = T / (2 * c);
      long long zs = (long long)2 * c * (T + 1);
      dim3 gd((c + BN - 1) / BN, (c + BM - 1) / BM, P);
      // tmp = Cinv @ off
      gemm_wmma<EPI_STORE, false, false, false><<<gd, 256, 0, stream>>>(
          Minv + (long long)c * (T + 1), T, Mb + (long long)c * T, T, tmp, T,
          c, c, c, zs, zs, (long long)c, nullptr, 0);
      // Minv[lower-left] = -tmp @ Ainv
      gemm_wmma<EPI_NEG, false, false, false><<<gd, 256, 0, stream>>>(
          tmp, T, Minv, T, Minv + (long long)c * T, T,
          c, c, c, (long long)c, zs, zs, nullptr, 0);
    }

    // btil = diag(beta) * Minv @ R      (triangular K clamps)
    gemm_wmma<EPI_RSCALE, false, true, true><<<gTT, 256, 0, stream>>>(Minv, T, R, T, btil, T, T, T, T, 0, 0, 0, beta + h, H);
    // A = Q K^T   (into R buffer) ;  tril(Q W^T) into Mb buffer
    gemm_wmma<EPI_STORE, true, false, false><<<gTT, 256, 0, stream>>>(qh, D, kh, D, R,  T, T, T, HDIM, 0, 0, 0, nullptr, 0);
    gemm_wmma<EPI_TRIL,  true, false, false><<<gTT, 256, 0, stream>>>(qh, D, wh, D, Mb, T, T, T, HDIM, 0, 0, 0, nullptr, 0);
    // A -= tril(QW) @ btil
    gemm_wmma<EPI_SUB, false, true, true><<<gTT, 256, 0, stream>>>(Mb, T, btil, T, R, T, T, T, T, 0, 0, 0, nullptr, 0);

    softmax_k<<<dim3(T), 256, 0, stream>>>(R, G, h, scale);

    // o_h = P @ V_h
    gemm_wmma<EPI_STORE, false, false, true><<<dim3(1, T / BM, 1), 256, 0, stream>>>(
        R, T, vh, D, o + h * HDIM, D, T, HDIM, T, 0, 0, 0, nullptr, 0);
  }

  // out = o @ Wo
  gemm_wmma<EPI_STORE, true, false, false><<<gproj, 256, 0, stream>>>(o, D, WoT, D, out, D, T, D, D, 0, 0, 0, nullptr, 0);
}
